// MLPPredictor_24352464569642
// MI455X (gfx1250) — compile-verified, matrix operand order checked by reading, not run on hardware
//
#include <hip/hip_runtime.h>

typedef _Float16 half_t;
typedef __attribute__((ext_vector_type(16))) _Float16 v16h;
typedef __attribute__((ext_vector_type(8)))  _Float16 v8h;
typedef __attribute__((ext_vector_type(4)))  _Float16 v4h;
typedef __attribute__((ext_vector_type(8)))  float    v8f;

#define HF 64            // H_FEATS
#define TILES_PER_WAVE 8

// ---------------------------------------------------------------------------
// Prep 1: convert node features f32 -> f16 (contiguous [node][64])
// ---------------------------------------------------------------------------
__global__ __launch_bounds__(256) void cvt_h_kernel(const float* __restrict__ in,
                                                    half_t* __restrict__ out,
                                                    long long n4) {
  long long i = (long long)blockIdx.x * blockDim.x + threadIdx.x;
  if (i < n4) {
    float4 v = ((const float4*)in)[i];
    v4h o;
    o[0] = (half_t)v.x; o[1] = (half_t)v.y; o[2] = (half_t)v.z; o[3] = (half_t)v.w;
    ((v4h*)out)[i] = o;
  }
}

// ---------------------------------------------------------------------------
// Prep 2: pack W1 [128,64] f32 into per-lane WMMA B-tile layout, f16.
// tile t = kt*4+nt, lane l, slot j at wB[t*512 + l*16 + j].
// K-within-32 for (lane,slot): (lane>>4)*8 + (j>>3)*16 + (j&7)
// ---------------------------------------------------------------------------
__global__ __launch_bounds__(256) void pack_w1_kernel(const float* __restrict__ W1,
                                                      half_t* __restrict__ wB) {
  int id = blockIdx.x * blockDim.x + threadIdx.x;   // 8192 total
  int j    = id & 15;
  int lane = (id >> 4) & 31;
  int tile = id >> 9;                                // 0..15
  int kt = tile >> 2, nt = tile & 3;
  int hf = lane >> 4;
  int k32 = hf * 8 + ((j >> 3) << 4) + (j & 7);
  int k = kt * 32 + k32;
  int n = nt * 16 + (lane & 15);
  wB[id] = (half_t)W1[k * HF + n];
}

// ---------------------------------------------------------------------------
// Gather one 16x128 A super-tile (4 k-steps of 16x32) for the wave's 16 edges.
// Each k-step has its OWN registers so all 8 loads issue as one clause.
// ---------------------------------------------------------------------------
template<bool PRE>
__device__ inline void load_a_tiles(const void* __restrict__ hptr,
                                    long long sN, long long dN,
                                    int fb_lo, v16h a[4]) {
#pragma unroll
  for (int kt = 0; kt < 4; ++kt) {
    const long long node = (kt < 2) ? sN : dN;
    const int fb = ((kt & 1) << 5) + fb_lo;
    if constexpr (PRE) {
      const half_t* p = (const half_t*)hptr + node * HF + fb;
      union { v16h v; v8h h[2]; } u;
      u.h[0] = *(const v8h*)p;          // K = fb .. fb+7
      u.h[1] = *(const v8h*)(p + 16);   // K = fb+16 .. fb+23
      a[kt] = u.v;
    } else {
      const float* p = (const float*)hptr + node * HF + fb;
      float f[16];
      *(float4*)(f + 0)  = *(const float4*)(p + 0);
      *(float4*)(f + 4)  = *(const float4*)(p + 4);
      *(float4*)(f + 8)  = *(const float4*)(p + 16);
      *(float4*)(f + 12) = *(const float4*)(p + 20);
#pragma unroll
      for (int j = 0; j < 16; ++j) a[kt][j] = (half_t)f[j];
    }
  }
}

// ---------------------------------------------------------------------------
// Main kernel: wave handles TILES_PER_WAVE tiles of 16 edges, software-
// pipelined: indices prefetched 2 tiles ahead, A gathers 1 tile ahead.
// ---------------------------------------------------------------------------
template<bool PRE>
__global__ __launch_bounds__(256) void edge_mlp_kernel(
    const void* __restrict__ hptr,
    const long long* __restrict__ src,
    const long long* __restrict__ dst,
    const half_t* __restrict__ wB,
    const float* __restrict__ b1v,
    const float* __restrict__ W2,
    const float* __restrict__ b2,
    float* __restrict__ out,
    long long nE, long long nTiles)
{
  const int lane = threadIdx.x & 31;
  const int m    = lane & 15;        // edge-in-tile / N index
  const int hf   = lane >> 4;        // half-wave
  const int fb_lo = hf << 3;
  const long long wave = (long long)blockIdx.x * (blockDim.x >> 5) + (threadIdx.x >> 5);
  const long long tile0 = wave * TILES_PER_WAVE;
  if (tile0 >= nTiles) return;

  // W1 B-tiles resident in registers for the whole wave (16 x v16h = 128 VGPRs)
  v16h B[16];
#pragma unroll
  for (int t = 0; t < 16; ++t)
    B[t] = *(const v16h*)(wB + t * 512 + lane * 16);

  const float bias[4] = { b1v[0*16 + m], b1v[1*16 + m], b1v[2*16 + m], b1v[3*16 + m] };
  const float w2l[4]  = { W2 [0*16 + m], W2 [1*16 + m], W2 [2*16 + m], W2 [3*16 + m] };
  const float bb2 = b2[0];

  // clamp edge id so prefetches past the end stay in-bounds (stores are masked)
  auto edge_of = [&](long long t) -> long long {
    long long tc = t < nTiles ? t : nTiles - 1;
    long long e  = tc * 16 + m;
    return e < nE ? e : nE - 1;
  };

  // ---- pipeline prologue: idx[t0], A[t0], idx[t0+1] ----
  long long e  = edge_of(tile0);
  long long sN = src[e], dN = dst[e];
  v16h a[4];
  load_a_tiles<PRE>(hptr, sN, dN, fb_lo, a);
  long long en  = edge_of(tile0 + 1);
  long long sNn = src[en], dNn = dst[en];

  for (int tt = 0; tt < TILES_PER_WAVE; ++tt) {
    const long long tileCur = tile0 + tt;
    if (tileCur >= nTiles) break;           // wave-uniform
    const long long e0 = tileCur * 16;

    // prefetch: A for tile+1, indices for tile+2 (issue before the WMMA block)
    v16h an[4];
    load_a_tiles<PRE>(hptr, sNn, dNn, fb_lo, an);
    const long long e2 = edge_of(tileCur + 2);
    const long long sN2 = src[e2], dN2 = dst[e2];

    // ---- 16 back-to-back WMMAs on the current A super-tile ----
    v8f c[4] = {};
#pragma unroll
    for (int kt = 0; kt < 4; ++kt)
#pragma unroll
      for (int nt = 0; nt < 4; ++nt)
        c[nt] = __builtin_amdgcn_wmma_f32_16x16x32_f16(
            false, a[kt], false, B[kt * 4 + nt], (short)0, c[nt], false, false);

    // ---- epilogue: relu(c + b1) . W2, reduced across the 16 N-lanes ----
    float acc[8];
#pragma unroll
    for (int g = 0; g < 8; ++g) acc[g] = 0.f;
#pragma unroll
    for (int nt = 0; nt < 4; ++nt) {
#pragma unroll
      for (int g = 0; g < 8; ++g) {
        float t = c[nt][g] + bias[nt];
        t = t > 0.f ? t : 0.f;
        acc[g] += t * w2l[nt];
      }
    }
#pragma unroll
    for (int off = 8; off >= 1; off >>= 1) {
#pragma unroll
      for (int g = 0; g < 8; ++g)
        acc[g] += __shfl_xor(acc[g], off, 16);
    }
    if (m == 0) {
#pragma unroll
      for (int g = 0; g < 8; ++g) {
        long long eo = e0 + (long long)hf * 8 + g;
        if (eo < nE) out[eo] = acc[g] + bb2;
      }
    }

    // ---- rotate pipeline registers ----
#pragma unroll
    for (int kt = 0; kt < 4; ++kt) a[kt] = an[kt];
    sNn = sN2; dNn = dN2;
  }
}

// ---------------------------------------------------------------------------
extern "C" void kernel_launch(void* const* d_in, const int* in_sizes, int n_in,
                              void* d_out, int out_size, void* d_ws, size_t ws_size,
                              hipStream_t stream) {
  const float*     h   = (const float*)d_in[0];
  const long long* src = (const long long*)d_in[1];
  const long long* dst = (const long long*)d_in[2];
  const float*     W1  = (const float*)d_in[3];
  const float*     b1  = (const float*)d_in[4];
  const float*     W2  = (const float*)d_in[5];
  const float*     b2  = (const float*)d_in[6];
  float*           out = (float*)d_out;

  const long long nNodes = in_sizes[0] / HF;
  const long long nE     = in_sizes[1];
  const long long nTiles = (nE + 15) / 16;

  // workspace layout: [packed W1 B-tiles: 16KB] [optional h16: nNodes*64*2 B]
  half_t* wB = (half_t*)d_ws;
  const size_t bBytes = 16 * 512 * sizeof(half_t);
  const size_t hOff   = (bBytes + 255) & ~(size_t)255;
  const size_t hBytes = (size_t)nNodes * HF * sizeof(half_t);
  const bool   pre    = (ws_size >= hOff + hBytes);
  half_t* h16 = pre ? (half_t*)((char*)d_ws + hOff) : nullptr;

  pack_w1_kernel<<<8192 / 256, 256, 0, stream>>>(W1, wB);

  if (pre) {
    long long n4 = nNodes * HF / 4;
    cvt_h_kernel<<<(int)((n4 + 255) / 256), 256, 0, stream>>>(h, h16, n4);
  }

  const long long waves  = (nTiles + TILES_PER_WAVE - 1) / TILES_PER_WAVE;
  const int       blocks = (int)((waves + 7) / 8);   // 8 waves / 256-thread block

  if (pre)
    edge_mlp_kernel<true ><<<blocks, 256, 0, stream>>>(h16, src, dst, wB, b1, W2, b2, out, nE, nTiles);
  else
    edge_mlp_kernel<false><<<blocks, 256, 0, stream>>>(h,   src, dst, wB, b1, W2, b2, out, nE, nTiles);
}
